// SelfAttention_73495480369388
// MI455X (gfx1250) — matrix operand order repaired and checked by reading, not compiled, not run on hardware
//
#include <hip/hip_runtime.h>

typedef __attribute__((ext_vector_type(16))) __bf16 v16bf;
typedef __attribute__((ext_vector_type(8)))  float  v8f;

#define B_  4
#define S_  2048
#define H_  16
#define D_  64
#define NCHUNK_ (S_ / 32)
// fold softmax scale (1/8) and log2(e) into Q so inner loop uses raw v_exp_f32
#define QSCALE_ 0.18033688011112042f   // 0.125 * log2(e)

// K-position permutation for the PV contraction (consistent between P and V):
//   k_pos 2c <-> key c ; k_pos 2c+1 <-> key c+16
// Mask applied at staging: masked keys get zeroed V rows / validity slots.

// single-instruction packed bf16 conversion (avoid scalarized union codegen)
__device__ __forceinline__ unsigned int pack_bf16(float lo, float hi) {
  unsigned int r;
  asm("v_cvt_pk_bf16_f32 %0, %1, %2" : "=v"(r) : "v"(lo), "v"(hi));
  return r;
}

union frag_u { v16bf v; uint4 q[2]; };

__global__ __launch_bounds__(128)
void fa_fwd_kernel(const float* __restrict__ qkv, const int* __restrict__ mask,
                   float* __restrict__ out) {
  const int b    = blockIdx.z;
  const int h    = blockIdx.y;
  const int tid  = threadIdx.x;
  const int wave = tid >> 5;
  const int lane = tid & 31;
  const int l16  = lane & 15;
  const int hi   = lane >> 4;
  const int q0   = blockIdx.x * 128 + wave * 32;  // first of 32 query rows for this wave

  // double-buffered staging + per-(wave,m-tile) P scratch
  __shared__ __align__(16) unsigned short k_lds[2][32 * 64];       // [key][d] bf16
  __shared__ __align__(16) unsigned short vb_lds[2][4 * 32 * 16];  // B-frag order, k-permuted, masked
  __shared__ __align__(16) unsigned short valid_lds[2][32];        // bf16 1.0/0.0 per k_pos
  __shared__ __align__(16) unsigned short p_lds[8 * 16 * 32];

  const size_t sS = (size_t)3 * H_ * D_;  // float stride between sequence positions
  const float* qb = qkv + (size_t)b * S_ * 3 * H_ * D_ + 0 * (H_ * D_) + (size_t)h * D_;
  const float* kb = qkv + (size_t)b * S_ * 3 * H_ * D_ + 1 * (H_ * D_) + (size_t)h * D_;
  const float* vb = qkv + (size_t)b * S_ * 3 * H_ * D_ + 2 * (H_ * D_) + (size_t)h * D_;
  const int*   mb = mask + (size_t)b * S_;

  // ---- Q: two m-tiles x two bf16 A fragments, pre-scaled by 0.125*log2e ----
  v16bf aA[2], aB[2];
#pragma unroll
  for (int mi = 0; mi < 2; ++mi) {
    const float* qrow = qb + (size_t)(q0 + mi * 16 + l16) * sS + 16 * hi;
#pragma unroll
    for (int j = 0; j < 8; ++j) aA[mi][j]     = (__bf16)(qrow[j]      * QSCALE_);
#pragma unroll
    for (int j = 0; j < 8; ++j) aA[mi][8 + j] = (__bf16)(qrow[8 + j] * QSCALE_);
#pragma unroll
    for (int j = 0; j < 8; ++j) aB[mi][j]     = (__bf16)(qrow[32 + j] * QSCALE_);
#pragma unroll
    for (int j = 0; j < 8; ++j) aB[mi][8 + j] = (__bf16)(qrow[40 + j] * QSCALE_);
  }

  v8f o[2][4], osum[2];
#pragma unroll
  for (int mi = 0; mi < 2; ++mi) {
#pragma unroll
    for (int e = 0; e < 8; ++e) osum[mi][e] = 0.0f;
#pragma unroll
    for (int t = 0; t < 4; ++t)
#pragma unroll
      for (int e = 0; e < 8; ++e) o[mi][t][e] = 0.0f;
  }

  // K staging: 4 threads per key row, 16 consecutive floats each
  const int srow = tid >> 2;
  const int sseg = tid & 3;
  // V staging: key pair (vk, vk+16) x 8 d-values -> adjacent k_pos slots (2vk, 2vk+1)
  const int vk   = tid & 15;
  const int voct = tid >> 4;
  const int vgrp = (vk >> 3) * 16;
  const int veE  = (2 * vk) & 15;

  // in-flight staging registers (pipeline stage: loads for chunk c+1 overlap compute of c)
  float kf[16], vf0[8], vf1[8];
  int   m0, m1;

  // lambda: issue global loads for chunk at c0 into registers
  auto load_stage = [&](int c0) {
    const float* kg = kb + (size_t)(c0 + srow) * sS + sseg * 16;
#pragma unroll
    for (int j = 0; j < 16; ++j) kf[j] = kg[j];
    const float* vg0 = vb + (size_t)(c0 + vk) * sS + voct * 8;
#pragma unroll
    for (int j = 0; j < 8; ++j) vf0[j] = vg0[j];
#pragma unroll
    for (int j = 0; j < 8; ++j) vf1[j] = vg0[16 * sS + j];
    m0 = mb[c0 + vk];
    m1 = mb[c0 + vk + 16];
    if (c0 + 64 < S_) {  // warm caches two chunks ahead (global_prefetch_b8)
      __builtin_prefetch(kg + 64 * sS, 0, 1);
      __builtin_prefetch(vg0 + 64 * sS, 0, 1);
    }
  };

  // lambda: convert + store staged registers into LDS buffer `sel`
  auto store_stage = [&](int sel) {
    uint4 kq0, kq1;
    unsigned int* kw = (unsigned int*)&kq0;
#pragma unroll
    for (int j = 0; j < 4; ++j) kw[j] = pack_bf16(kf[2 * j], kf[2 * j + 1]);
    kw = (unsigned int*)&kq1;
#pragma unroll
    for (int j = 0; j < 4; ++j) kw[j] = pack_bf16(kf[8 + 2 * j], kf[9 + 2 * j]);
    uint4* kd = (uint4*)(&k_lds[sel][0] + srow * 64 + sseg * 16);
    kd[0] = kq0;
    kd[1] = kq1;

    const unsigned int mw = (m0 ? 0x0000FFFFu : 0u) | (m1 ? 0xFFFF0000u : 0u);
    *(unsigned int*)(&valid_lds[sel][0] + 2 * vk) = 0x3F803F80u & mw;
#pragma unroll
    for (int j = 0; j < 8; ++j) {
      const int d  = voct * 8 + j;
      const int ti = d >> 4;
      const int li = d & 15;
      unsigned int* dst = (unsigned int*)(&vb_lds[sel][0] + ((ti * 32 + li + vgrp) << 4) + veE);
      *dst = pack_bf16(vf0[j], vf1[j]) & mw;   // k_pos slots 2vk, 2vk+1
    }
  };

  load_stage(0);

  for (int c = 0; c < NCHUNK_; ++c) {
    const int sel = c & 1;
    store_stage(sel);          // waits on chunk c's global loads
    __syncthreads();           // single barrier per chunk (see pipeline proof)
    if (c + 1 < NCHUNK_) load_stage((c + 1) * 32);  // overlap next chunk's loads with compute

    // ---- K fragments (8 x ds_load_b128), scores + exp per m-tile ----
    const uint4* kb4 = (const uint4*)&k_lds[sel][0];
    const int i0 = l16 * 8 + 2 * hi;
    const int i1 = (16 + l16) * 8 + 2 * hi;
    frag_u fk0, fk1, fk2, fk3;
    fk0.q[0] = kb4[i0];     fk0.q[1] = kb4[i0 + 1];
    fk1.q[0] = kb4[i0 + 4]; fk1.q[1] = kb4[i0 + 5];
    fk2.q[0] = kb4[i1];     fk2.q[1] = kb4[i1 + 1];
    fk3.q[0] = kb4[i1 + 4]; fk3.q[1] = kb4[i1 + 5];

#pragma unroll
    for (int mi = 0; mi < 2; ++mi) {
      v8f s0, s1;
#pragma unroll
      for (int e = 0; e < 8; ++e) { s0[e] = 0.0f; s1[e] = 0.0f; }
      s0 = __builtin_amdgcn_wmma_f32_16x16x32_bf16(false, aA[mi], false, fk0.v, (short)0, s0, false, false);
      s0 = __builtin_amdgcn_wmma_f32_16x16x32_bf16(false, aB[mi], false, fk1.v, (short)0, s0, false, false);
      s1 = __builtin_amdgcn_wmma_f32_16x16x32_bf16(false, aA[mi], false, fk2.v, (short)0, s1, false, false);
      s1 = __builtin_amdgcn_wmma_f32_16x16x32_bf16(false, aB[mi], false, fk3.v, (short)0, s1, false, false);

      unsigned short* pls = p_lds + (wave * 2 + mi) * (16 * 32);
#pragma unroll
      for (int r = 0; r < 8; ++r) {
        const float e0 = __builtin_amdgcn_exp2f(s0[r]);
        const float e1 = __builtin_amdgcn_exp2f(s1[r]);
        unsigned int* dst = (unsigned int*)(pls + (r + 8 * hi) * 32 + 2 * l16);
        *dst = pack_bf16(e0, e1);   // k_pos 2*l16 (key l16), 2*l16+1 (key l16+16)
      }
    }

    // ---- V / validity / P fragments, then PV + row-sum WMMAs ----
    const uint4* vb4 = (const uint4*)&vb_lds[sel][0];
    frag_u fv[4], fvalid, pa[2];
#pragma unroll
    for (int t = 0; t < 4; ++t) {
      const int vi = (t * 32 + lane) * 2;
      fv[t].q[0] = vb4[vi];
      fv[t].q[1] = vb4[vi + 1];
    }
    fvalid.q[0] = ((const uint4*)&valid_lds[sel][0])[2 * hi];
    fvalid.q[1] = ((const uint4*)&valid_lds[sel][0])[2 * hi + 1];
#pragma unroll
    for (int mi = 0; mi < 2; ++mi) {
      const uint4* pb4 = (const uint4*)(p_lds + (wave * 2 + mi) * (16 * 32) + l16 * 32);
      pa[mi].q[0] = pb4[2 * hi];
      pa[mi].q[1] = pb4[2 * hi + 1];
    }
#pragma unroll
    for (int mi = 0; mi < 2; ++mi) {
#pragma unroll
      for (int t = 0; t < 4; ++t)
        o[mi][t] = __builtin_amdgcn_wmma_f32_16x16x32_bf16(false, pa[mi].v, false, fv[t].v,
                                                           (short)0, o[mi][t], false, false);
      osum[mi] = __builtin_amdgcn_wmma_f32_16x16x32_bf16(false, pa[mi].v, false, fvalid.v,
                                                         (short)0, osum[mi], false, false);
    }
  }

  // ---- normalize by row sums (broadcast across columns) and store ----
  float* ob = out + (size_t)b * S_ * H_ * D_ + (size_t)h * D_;
#pragma unroll
  for (int mi = 0; mi < 2; ++mi) {
#pragma unroll
    for (int r = 0; r < 8; ++r) {
      const float inv = __builtin_amdgcn_rcpf(osum[mi][r]);
      float* orow = ob + (size_t)(q0 + mi * 16 + r + 8 * hi) * (H_ * D_);
#pragma unroll
      for (int t = 0; t < 4; ++t) orow[t * 16 + l16] = o[mi][t][r] * inv;
    }
  }
}

extern "C" void kernel_launch(void* const* d_in, const int* in_sizes, int n_in,
                              void* d_out, int out_size, void* d_ws, size_t ws_size,
                              hipStream_t stream) {
  const float* qkv  = (const float*)d_in[0];
  const int*   mask = (const int*)d_in[1];
  float*       out  = (float*)d_out;
  (void)in_sizes; (void)n_in; (void)out_size; (void)d_ws; (void)ws_size;

  dim3 grid(S_ / 128, H_, B_);   // (query tiles of 128 rows, heads, batch)
  fa_fwd_kernel<<<grid, 128, 0, stream>>>(qkv, mask, out);
}
